// JpegProxy_53420803227686
// MI455X (gfx1250) — matrix operand order repaired and checked by reading, not compiled, not run on hardware
//
#include <hip/hip_runtime.h>
#include <hip/hip_bf16.h>

typedef __attribute__((ext_vector_type(16))) _Float16 v16h;
typedef __attribute__((ext_vector_type(8)))  _Float16 v8h;
typedef __attribute__((ext_vector_type(8)))  float    v8f;

union FragH { v16h v; v8h h[2]; };

#define JPG_PI 3.14159265358979323846f

static __device__ __forceinline__ v8f wmma_f16(v16h a, v16h b, v8f c) {
    return __builtin_amdgcn_wmma_f32_16x16x32_f16(false, a, false, b, (short)0, c, false, false);
}

// ---------------------------------------------------------------------------
// Kernel 1: RGB -> YUV (x255, +128 chroma offset) + 2x2 chroma downsample.
// One thread per 2x2 quad; float2 loads/stores keep everything coalesced.
// ---------------------------------------------------------------------------
__global__ __launch_bounds__(256) void k_rgb2yuv(const float* __restrict__ img,
                                                 float* __restrict__ Yp,
                                                 float* __restrict__ Up,
                                                 float* __restrict__ Vp)
{
    int t  = blockIdx.x * 256 + threadIdx.x;   // 16 * 256 * 256 quads
    int b  = t >> 16;
    int qy = (t >> 8) & 255;
    int qx = t & 255;

    size_t base = (size_t)b * 3 * 512 * 512;
    const float* Rp = img + base;
    const float* Gp = img + base + 512 * 512;
    const float* Bp = img + base + 2 * 512 * 512;
    float* Yb = Yp + (size_t)b * 512 * 512;

    float us = 0.f, vs = 0.f;
#pragma unroll
    for (int dy = 0; dy < 2; ++dy) {
        int y = qy * 2 + dy;
        size_t off = (size_t)y * 512 + qx * 2;
        float2 r2 = *(const float2*)(Rp + off);
        float2 g2 = *(const float2*)(Gp + off);
        float2 b2 = *(const float2*)(Bp + off);
        float yo[2];
#pragma unroll
        for (int dx = 0; dx < 2; ++dx) {
            float r  = (dx ? r2.y : r2.x) * 255.f;
            float g  = (dx ? g2.y : g2.x) * 255.f;
            float bb = (dx ? b2.y : b2.x) * 255.f;
            yo[dx] = 0.299f * r + 0.587f * g + 0.114f * bb;
            us += -0.168736f * r - 0.331264f * g + 0.5f      * bb;
            vs +=  0.5f      * r - 0.418688f * g - 0.081312f * bb;
        }
        float2 yw; yw.x = yo[0]; yw.y = yo[1];
        *(float2*)(Yb + off) = yw;
    }
    size_t coff = ((size_t)b * 256 + qy) * 256 + qx;
    Up[coff] = us * 0.25f + 128.0f;
    Vp[coff] = vs * 0.25f + 128.0f;
}

// ---------------------------------------------------------------------------
// Kernel 2: per-8x8-block DCT -> quantize -> write q -> dequantize -> IDCT,
// reconstruction written IN PLACE over the source plane.
// Workgroup = 256 threads = 8 waves; each wave owns 16 blocks and runs
// C[16x64] = A[16x64] x M[64x64] as 4 N-tiles x 2 K-chunks of
// v_wmma_f32_16x16x32_f16 (and the same for the inverse with M^T).
// ---------------------------------------------------------------------------
__global__ __launch_bounds__(256) void k_dct(float* plane,            // src & rec (in place)
                                             float* __restrict__ qOut,
                                             const float* __restrict__ qt,
                                             int Hc, int Wc, int nbShift, int bwShift)
{
    __shared__ float sD[64];
    __shared__ __align__(16) _Float16 sM [64 * 72];   // M[i][j]  (i=spatial, j=coeff) -> inverse B
    __shared__ __align__(16) _Float16 sMT[64 * 72];   // M^T[j][i]                    -> forward B
    __shared__ __align__(16) _Float16 sPix[128 * 72]; // staged pixels / dequantized coeffs

    const int tid = threadIdx.x;

    // Build 8x8 DCT basis D, then kron(D,D) into LDS (f16, both orientations).
    if (tid < 64) {
        int i = tid >> 3, j = tid & 7;
        float v = __cosf((float)(2 * i + 1) * (float)j * (JPG_PI / 16.0f)) * 0.5f;
        if (j == 0) v *= 0.70710678118654752f;
        sD[tid] = v;
    }
    __syncthreads();
    for (int idx = tid; idx < 4096; idx += 256) {
        int i = idx >> 6, j = idx & 63;
        float v = sD[((i >> 3) << 3) | (j >> 3)] * sD[((i & 7) << 3) | (j & 7)];
        _Float16 h = (_Float16)v;
        sM [i * 72 + j] = h;
        sMT[j * 72 + i] = h;
    }

    // Stage 128 blocks (pix - 128 -> f16) into LDS, [block][64], stride 72 halves.
    const int wgBase = blockIdx.x << 7;
    const int bw     = 1 << bwShift;
    const int nbMask = (1 << nbShift) - 1;
    for (int idx = tid; idx < 128 * 64; idx += 256) {
        int bl = idx >> 6, k = idx & 63;
        int g  = wgBase + bl;
        int b  = g >> nbShift;
        int bi = g & nbMask;
        int by = bi >> bwShift, bx = bi & (bw - 1);
        int y  = (by << 3) + (k >> 3), x = (bx << 3) + (k & 7);
        float p = plane[((size_t)b * Hc + y) * Wc + x] - 128.0f;
        sPix[bl * 72 + k] = (_Float16)p;
    }
    __syncthreads();

    const int w       = tid >> 5;     // wave id 0..7
    const int lane    = tid & 31;
    const int n0      = lane & 15;
    const int hi      = lane >> 4;
    const int rowBase = w << 4;       // this wave's first local block

    // Quantization table values: coefficient index depends only on (tile, lane).
    // Hoist the reciprocal so quantization is a multiply (avoids 32 IEEE
    // v_div_scale/v_rcp expansions per thread -- the dominant VALU cost).
    float qv[4], iq[4];
#pragma unroll
    for (int t = 0; t < 4; ++t) {
        qv[t] = qt[t * 16 + n0];
        iq[t] = 1.0f / qv[t];
    }

    // A fragments (pixels), 2 K-chunks of 32.
    FragH A[2];
#pragma unroll
    for (int c = 0; c < 2; ++c) {
        const _Float16* p = &sPix[(rowBase + n0) * 72 + c * 32 + hi * 8];
        A[c].h[0] = *(const v8h*)p;
        A[c].h[1] = *(const v8h*)(p + 16);
    }

    // Forward DCT: C[t] = A x M (4 N-tiles).
    v8f C[4];
#pragma unroll
    for (int t = 0; t < 4; ++t) {
        v8f acc = {0.f, 0.f, 0.f, 0.f, 0.f, 0.f, 0.f, 0.f};
#pragma unroll
        for (int c = 0; c < 2; ++c) {
            FragH Bf;
            const _Float16* p = &sMT[(t * 16 + n0) * 72 + c * 32 + hi * 8];
            Bf.h[0] = *(const v8h*)p;
            Bf.h[1] = *(const v8h*)(p + 16);
            acc = wmma_f16(A[c].v, Bf.v, acc);
        }
        C[t] = acc;
    }

    // Quantize (round-half-even, matching jnp.round), emit q, stash dequantized
    // coeffs back into this wave's own LDS region (per-wave DS ordering).
    #pragma unroll
    for (int t = 0; t < 4; ++t) {
        float qscale = qv[t];
        float qinv   = iq[t];
        int n = t * 16 + n0;
#pragma unroll
        for (int r = 0; r < 8; ++r) {
            int m = r + (hi << 3);
            int g = wgBase + rowBase + m;
            float qc = rintf(C[t][r] * qinv);
            qOut[(size_t)g * 64 + n] = qc;
            sPix[(rowBase + m) * 72 + n] = (_Float16)(qc * qscale);
        }
    }

    // Inverse DCT: rec = Q x M^T, +128, write back to the plane in place.
    FragH A2[2];
#pragma unroll
    for (int c = 0; c < 2; ++c) {
        const _Float16* p = &sPix[(rowBase + n0) * 72 + c * 32 + hi * 8];
        A2[c].h[0] = *(const v8h*)p;
        A2[c].h[1] = *(const v8h*)(p + 16);
    }
#pragma unroll
    for (int t = 0; t < 4; ++t) {
        v8f acc = {0.f, 0.f, 0.f, 0.f, 0.f, 0.f, 0.f, 0.f};
#pragma unroll
        for (int c = 0; c < 2; ++c) {
            FragH Bi;
            const _Float16* p = &sM[(t * 16 + n0) * 72 + c * 32 + hi * 8];
            Bi.h[0] = *(const v8h*)p;
            Bi.h[1] = *(const v8h*)(p + 16);
            acc = wmma_f16(A2[c].v, Bi.v, acc);
        }
        int k  = t * 16 + n0;              // pixel index within block
        int ky = k >> 3, kx = k & 7;
#pragma unroll
        for (int r = 0; r < 8; ++r) {
            int m  = r + (hi << 3);
            int g  = wgBase + rowBase + m;
            int b  = g >> nbShift;
            int bi = g & nbMask;
            int by = bi >> bwShift, bx = bi & (bw - 1);
            plane[((size_t)b * Hc + (by << 3) + ky) * Wc + (bx << 3) + kx] = acc[r] + 128.0f;
        }
    }
}

// ---------------------------------------------------------------------------
// Kernel 3: bilinear chroma upsample (half-pixel centers, edge clamp -- matches
// jax.image.resize 'bilinear' for 2x upsample) + YUV->RGB + clip + /255, NCHW.
// ---------------------------------------------------------------------------
__global__ __launch_bounds__(256) void k_final(const float* __restrict__ Yp,
                                               const float* __restrict__ Up,
                                               const float* __restrict__ Vp,
                                               float* __restrict__ out)
{
    int t = blockIdx.x * 256 + threadIdx.x;   // 16 * 512 * 512
    int b = t >> 18;
    int y = (t >> 9) & 511;
    int x = t & 511;

    float yv = Yp[((size_t)b * 512 + y) * 512 + x];

    float sx = x * 0.5f - 0.25f;
    float sy = y * 0.5f - 0.25f;
    float fx = floorf(sx), fy = floorf(sy);
    float wx = sx - fx,    wy = sy - fy;
    int x0 = max((int)fx, 0), x1 = min((int)fx + 1, 255);
    int y0 = max((int)fy, 0), y1 = min((int)fy + 1, 255);

    const float* Ub = Up + (size_t)b * 256 * 256;
    const float* Vb = Vp + (size_t)b * 256 * 256;
    float u00 = Ub[y0 * 256 + x0], u01 = Ub[y0 * 256 + x1];
    float u10 = Ub[y1 * 256 + x0], u11 = Ub[y1 * 256 + x1];
    float v00 = Vb[y0 * 256 + x0], v01 = Vb[y0 * 256 + x1];
    float v10 = Vb[y1 * 256 + x0], v11 = Vb[y1 * 256 + x1];
    float uu = (1.f - wy) * ((1.f - wx) * u00 + wx * u01) + wy * ((1.f - wx) * u10 + wx * u11);
    float vv = (1.f - wy) * ((1.f - wx) * v00 + wx * v01) + wy * ((1.f - wx) * v10 + wx * v11);

    float up = uu - 128.0f, vp = vv - 128.0f;
    float r = yv + 1.402f    * vp;
    float g = yv - 0.344136f * up - 0.714136f * vp;
    float bb = yv + 1.772f   * up;

    const float s = 1.0f / 255.0f;
    r  = fminf(fmaxf(r,  0.f), 255.f) * s;
    g  = fminf(fmaxf(g,  0.f), 255.f) * s;
    bb = fminf(fmaxf(bb, 0.f), 255.f) * s;

    size_t base = ((size_t)b * 3) * 512 * 512 + (size_t)y * 512 + x;
    out[base]                  = r;
    out[base + 512 * 512]      = g;
    out[base + 2 * 512 * 512]  = bb;
}

// ---------------------------------------------------------------------------
extern "C" void kernel_launch(void* const* d_in, const int* in_sizes, int n_in,
                              void* d_out, int out_size, void* d_ws, size_t ws_size,
                              hipStream_t stream)
{
    (void)in_sizes; (void)n_in; (void)out_size; (void)ws_size;

    const float* image  = (const float*)d_in[0];   // (16, 3, 512, 512)
    const float* luma   = (const float*)d_in[1];   // (64,)
    const float* chroma = (const float*)d_in[2];   // (64,)
    float* out = (float*)d_out;

    const size_t B = 16, H = 512, W = 512, Hc = 256, Wc = 256;

    float* Yp = (float*)d_ws;                 // 16 * 512 * 512 f32
    float* Up = Yp + B * H * W;               // 16 * 256 * 256 f32
    float* Vp = Up + B * Hc * Wc;             // 16 * 256 * 256 f32

    float* rgbOut = out;                      // 16*3*512*512
    float* qY = out + B * 3 * H * W;          // 16*4096*64
    float* qU = qY + B * 4096 * 64;           // 16*1024*64
    float* qV = qU + B * 1024 * 64;           // 16*1024*64

    // 1) color transform + chroma downsample
    k_rgb2yuv<<<4096, 256, 0, stream>>>(image, Yp, Up, Vp);

    // 2) DCT/quant/IDCT per channel (rec in place over plane)
    //    Y: 16*4096 blocks -> 512 WGs;  U/V: 16*1024 blocks -> 128 WGs each
    k_dct<<<512, 256, 0, stream>>>(Yp, qY, luma,   (int)H,  (int)W,  12, 6);
    k_dct<<<128, 256, 0, stream>>>(Up, qU, chroma, (int)Hc, (int)Wc, 10, 5);
    k_dct<<<128, 256, 0, stream>>>(Vp, qV, chroma, (int)Hc, (int)Wc, 10, 5);

    // 3) chroma upsample + YUV->RGB + clip
    k_final<<<16384, 256, 0, stream>>>(Yp, Up, Vp, rgbOut);
}
